// Net_3547642986644
// MI455X (gfx1250) — matrix operand order repaired and checked by reading, not compiled, not run on hardware
//
#include <hip/hip_runtime.h>

// ---------------------------------------------------------------------------
// GAT (2-layer) forward for MI455X / gfx1250.
// GEMMs run on v_wmma_f32_16x16x32_bf16 (f32 accum, bf16 multiplicands
// converted on the fly); attention softmax/scatter in f32 with L2 atomics.
// ---------------------------------------------------------------------------

#define NN    100000
#define EE    3200000
#define ENTOT (EE + NN)      // edges + self loops
#define FIN   1433
#define KPAD  1440           // 45 * 32
#define NSLOPE 0.2f

typedef __attribute__((ext_vector_type(16))) __bf16         v16bf;
typedef __attribute__((ext_vector_type(8)))  float          v8f;
typedef __attribute__((ext_vector_type(8)))  unsigned short ushort8;

union FragBF {
    v16bf          v;
    ushort8        h[2];
    unsigned short u[16];
};

// f32 -> bf16 (round to nearest even), as raw ushort bits
__device__ __forceinline__ unsigned short f2bf(float f) {
    unsigned u = __float_as_uint(f);
    u = (u + 0x7FFFu + ((u >> 16) & 1u)) >> 16;
    return (unsigned short)u;
}

// order-preserving float <-> uint key (for atomicMax on floats)
__device__ __forceinline__ unsigned fkey(float f) {
    unsigned u = __float_as_uint(f);
    return (u & 0x80000000u) ? ~u : (u | 0x80000000u);
}
__device__ __forceinline__ float funkey(unsigned k) {
    return __uint_as_float((k & 0x80000000u) ? (k & 0x7FFFFFFFu) : ~k);
}

__device__ __forceinline__ void pack4(float4 q, unsigned short* d) {
    d[0] = f2bf(q.x); d[1] = f2bf(q.y); d[2] = f2bf(q.z); d[3] = f2bf(q.w);
}

// ---------------------------------------------------------------------------
// Workspace layout (bytes)
// ---------------------------------------------------------------------------
constexpr size_t OFF_W1BF = 0;          // 64*KPAD bf16 (W1^T)   = 184320
constexpr size_t OFF_W2BF = 184320;     // 16*64 bf16 (B^T, pad) =   2048
constexpr size_t OFF_H1   = 186368;     // N*64 f32  (x @ W1)
constexpr size_t OFF_AS1  = 25786368;   // N*8  f32
constexpr size_t OFF_AD1  = 28986368;   // N*8  f32
constexpr size_t OFF_MAX1 = 32186368;   // N*8  u32 (ordered key)
constexpr size_t OFF_DEN1 = 35386368;   // N*8  f32
constexpr size_t OFF_ACC1 = 38586368;   // N*64 f32
constexpr size_t OFF_HE   = 64186368;   // N*64 f32  elu(out1+b1)
constexpr size_t OFF_H2   = 89786368;   // N*8  f32  (helu @ W2, stride 8)
constexpr size_t OFF_AS2  = 92986368;   // N f32
constexpr size_t OFF_AD2  = 93386368;   // N f32
constexpr size_t OFF_MAX2 = 93786368;   // N u32
constexpr size_t OFF_DEN2 = 94186368;   // N f32
constexpr size_t OFF_ACC2 = 94586368;   // N*8 f32

// ---------------------------------------------------------------------------
// Weight conversion
// ---------------------------------------------------------------------------
// W1: [1433,64] f32 -> w1bfT: [64][KPAD] bf16 (transposed, K zero-padded)
__global__ void k_conv_w1(const float* __restrict__ W1, unsigned short* __restrict__ w1bfT) {
    int gid = blockIdx.x * 256 + threadIdx.x;          // 64*KPAD
    if (gid >= 64 * KPAD) return;
    int n = gid / KPAD, k = gid - n * KPAD;
    float v = (k < FIN) ? W1[(size_t)k * 64 + n] : 0.0f;
    w1bfT[gid] = f2bf(v);
}

// W2: [64,7] f32 -> w2bfT: [16][64] bf16 (cols 7..15 zero), transposed
__global__ void k_conv_w2(const float* __restrict__ W2, unsigned short* __restrict__ w2bfT) {
    int gid = blockIdx.x * 256 + threadIdx.x;          // 16*64
    if (gid >= 16 * 64) return;
    int n = gid >> 6, k = gid & 63;
    float v = (n < 7) ? W2[(size_t)k * 7 + n] : 0.0f;
    w2bfT[gid] = f2bf(v);
}

__global__ void k_fill(unsigned* __restrict__ p, unsigned v, int n) {
    int gid = blockIdx.x * 256 + threadIdx.x;
    if (gid < n) p[gid] = v;
}

// ---------------------------------------------------------------------------
// GEMM1: h1 = x @ W1   [N,1433]x[1433,64] -> [N,64]  (no bias here)
// Block = 256 thr = 8 waves; block tile 32x64; wave tile 16x16.
// ---------------------------------------------------------------------------
__global__ void __launch_bounds__(256)
k_gemm1(const float* __restrict__ x, const unsigned short* __restrict__ w1bfT,
        float* __restrict__ h1) {
    __shared__ __align__(16) unsigned short ldsA[32 * 40];  // 32 rows x 32k (+8 pad)
    __shared__ __align__(16) unsigned short ldsB[64 * 40];  // B^T: 64 cols x 32k (+8 pad)

    const int t   = threadIdx.x;
    const int m0  = blockIdx.x * 32;
    const int wid = t >> 5, lane = t & 31;
    const int wr  = wid >> 2, wc = wid & 3;
    const int l15 = lane & 15, lh = lane >> 4;

    const int arow = t >> 3;            // A staging: row 0..31
    const int akb  = (t & 7) * 4;       // A staging: k base 0..28
    const int bn   = t >> 2;            // B staging: col 0..63
    const int bko  = (t & 3) * 8;       // B staging: k base 0,8,16,24

    const size_t abase = (size_t)(m0 + arow) * FIN + akb;   // + k0 at use
    const size_t bbase = (size_t)bn * KPAD + bko;           // + k0 at use

    v8f acc = {};

    for (int k0 = 0; k0 < KPAD; k0 += 32) {
        // ---- stage A tile (f32 -> bf16, packed b64 DS store) ----
        {
            const float* ap = x + abase + k0;
            __builtin_prefetch(ap + 32, 0, 1);              // next K-tile hint
            float v0, v1, v2, v3;
            if (k0 + 32 <= FIN) {                           // uniform fast path
                v0 = ap[0]; v1 = ap[1]; v2 = ap[2]; v3 = ap[3];
            } else {                                        // tail step only
                int kk = k0 + akb;
                v0 = (kk + 0 < FIN) ? ap[0] : 0.0f;
                v1 = (kk + 1 < FIN) ? ap[1] : 0.0f;
                v2 = (kk + 2 < FIN) ? ap[2] : 0.0f;
                v3 = (kk + 3 < FIN) ? ap[3] : 0.0f;
            }
            uint2 pk;
            pk.x = (unsigned)f2bf(v0) | ((unsigned)f2bf(v1) << 16);
            pk.y = (unsigned)f2bf(v2) | ((unsigned)f2bf(v3) << 16);
            *(uint2*)(ldsA + arow * 40 + akb) = pk;
        }
        // ---- stage B^T tile: one b128 load + one b128 DS store per thread ----
        *(ushort8*)(ldsB + bn * 40 + bko) =
            *(const ushort8*)(w1bfT + bbase + k0);
        __syncthreads();

        FragBF fa, fb;
        const unsigned short* pa = ldsA + (wr * 16 + l15) * 40 + lh * 16;
        fa.h[0] = *(const ushort8*)(pa);        // K = lh*16 + 0..7
        fa.h[1] = *(const ushort8*)(pa + 8);    // K = lh*16 + 8..15
        const unsigned short* pb = ldsB + (wc * 16 + l15) * 40 + lh * 16;
        fb.h[0] = *(const ushort8*)(pb);        // K = lh*16 + 0..7
        fb.h[1] = *(const ushort8*)(pb + 8);    // K = lh*16 + 8..15

        acc = __builtin_amdgcn_wmma_f32_16x16x32_bf16(
                  false, fa.v, false, fb.v, (short)0, acc, false, false);
        __syncthreads();
    }

    const int col = wc * 16 + l15;
    #pragma unroll
    for (int i = 0; i < 8; ++i) {
        int m = m0 + wr * 16 + i + 8 * lh;      // C/D layout: M = i (+8 for hi lanes)
        h1[(size_t)m * 64 + col] = acc[i];
    }
}

// ---------------------------------------------------------------------------
// GEMM2: h2 = helu @ W2   [N,64]x[64,16pad] -> [N,7] (stride 8)
// One 16-row tile per wave; K=64 = two bf16 WMMA steps.
// ---------------------------------------------------------------------------
__global__ void __launch_bounds__(256)
k_gemm2(const float* __restrict__ helu, const unsigned short* __restrict__ w2bfT,
        float* __restrict__ h2p) {
    const int wid  = threadIdx.x >> 5, lane = threadIdx.x & 31;
    const int tile = blockIdx.x * 8 + wid;
    if (tile >= NN / 16) return;                 // wave-uniform branch
    const int l15 = lane & 15, lh = lane >> 4;
    const int row = tile * 16 + l15;

    v8f acc = {};
    #pragma unroll
    for (int ks = 0; ks < 2; ++ks) {
        FragBF fa, fb;
        const float* ar = helu + (size_t)row * 64 + ks * 32 + lh * 8;
        float4 q0 = *(const float4*)(ar);
        float4 q1 = *(const float4*)(ar + 4);
        float4 q2 = *(const float4*)(ar + 16);
        float4 q3 = *(const float4*)(ar + 20);
        pack4(q0, &fa.u[0]); pack4(q1, &fa.u[4]);
        pack4(q2, &fa.u[8]); pack4(q3, &fa.u[12]);
        const ushort8* bp = (const ushort8*)(w2bfT + l15 * 64 + ks * 32 + lh * 8);
        fb.h[0] = bp[0]; fb.h[1] = bp[2];
        acc = __builtin_amdgcn_wmma_f32_16x16x32_bf16(
                  false, fa.v, false, fb.v, (short)0, acc, false, false);
    }
    if (l15 < 7) {
        #pragma unroll
        for (int i = 0; i < 8; ++i) {
            int m = tile * 16 + i + 8 * lh;
            h2p[(size_t)m * 8 + l15] = acc[i];
        }
    }
}

// ---------------------------------------------------------------------------
// Attention source/dest dot products
// ---------------------------------------------------------------------------
__global__ void k_attdot1(const float* __restrict__ h1,
                          const float* __restrict__ attS, const float* __restrict__ attD,
                          float* __restrict__ aS, float* __restrict__ aD) {
    int gid = blockIdx.x * 256 + threadIdx.x;    // N*8
    if (gid >= NN * 8) return;
    int n = gid >> 3, h = gid & 7;
    const float* r = h1 + ((size_t)n << 6) + (h << 3);
    float s = 0.f, d = 0.f;
    #pragma unroll
    for (int c = 0; c < 8; ++c) {
        s += r[c] * attS[(h << 3) + c];
        d += r[c] * attD[(h << 3) + c];
    }
    aS[gid] = s; aD[gid] = d;
}

__global__ void k_attdot2(const float* __restrict__ h2p,
                          const float* __restrict__ attS, const float* __restrict__ attD,
                          float* __restrict__ aS, float* __restrict__ aD) {
    int n = blockIdx.x * 256 + threadIdx.x;
    if (n >= NN) return;
    const float* r = h2p + (size_t)n * 8;
    float s = 0.f, d = 0.f;
    #pragma unroll
    for (int c = 0; c < 7; ++c) { s += r[c] * attS[c]; d += r[c] * attD[c]; }
    aS[n] = s; aD[n] = d;
}

// ---------------------------------------------------------------------------
// Edge phase: pass 1 = segment max, pass 2 = weighted scatter-add
// ---------------------------------------------------------------------------
template <int H>
__global__ void k_edge_max(const long long* __restrict__ ei,
                           const float* __restrict__ aS, const float* __restrict__ aD,
                           unsigned* __restrict__ maxb) {
    int idx = blockIdx.x * 256 + threadIdx.x;    // ENTOT
    if (idx >= ENTOT) return;
    int s, d;
    if (idx < EE) { s = (int)ei[idx]; d = (int)ei[EE + idx]; }
    else          { s = idx - EE; d = s; }
    #pragma unroll
    for (int h = 0; h < H; ++h) {
        float e  = aS[(size_t)s * H + h] + aD[(size_t)d * H + h];
        float lr = (e > 0.f) ? e : NSLOPE * e;
        atomicMax(maxb + (size_t)d * H + h, fkey(lr));
    }
}

template <int H, int C, int S>
__global__ void k_edge_acc(const long long* __restrict__ ei,
                           const float* __restrict__ aS, const float* __restrict__ aD,
                           const unsigned* __restrict__ maxb,
                           float* __restrict__ den,
                           const float* __restrict__ feat, float* __restrict__ acc) {
    int gid = blockIdx.x * 256 + threadIdx.x;    // ENTOT * H
    if (gid >= ENTOT * H) return;
    int idx = gid / H;
    int h   = gid - idx * H;
    int s, d;
    if (idx < EE) { s = (int)ei[idx]; d = (int)ei[EE + idx]; }
    else          { s = idx - EE; d = s; }
    float e  = aS[(size_t)s * H + h] + aD[(size_t)d * H + h];
    float lr = (e > 0.f) ? e : NSLOPE * e;
    float m  = funkey(maxb[(size_t)d * H + h]);
    float w  = __expf(lr - m);
    atomicAdd(den + (size_t)d * H + h, w);
    const float* fs = feat + (size_t)s * S + h * C;
    float*       fd = acc  + (size_t)d * S + h * C;
    #pragma unroll
    for (int c = 0; c < C; ++c) atomicAdd(fd + c, w * fs[c]);
}

// ---------------------------------------------------------------------------
// Finalize layer1: out1/denom + b1, then ELU -> helu
// ---------------------------------------------------------------------------
__global__ void k_final1(const float* __restrict__ acc1, const float* __restrict__ den1,
                         const float* __restrict__ b1, float* __restrict__ helu) {
    int gid = blockIdx.x * 256 + threadIdx.x;    // N*64
    if (gid >= NN * 64) return;
    int n = gid >> 6, hc = gid & 63, h = hc >> 3;
    float v = acc1[gid] / den1[(n << 3) + h] + b1[hc];
    helu[gid] = (v > 0.f) ? v : (__expf(v) - 1.0f);
}

// Finalize layer2: out2/denom + b2, then log_softmax over 7 classes
__global__ void k_final2(const float* __restrict__ acc2, const float* __restrict__ den2,
                         const float* __restrict__ b2, float* __restrict__ out) {
    int n = blockIdx.x * 256 + threadIdx.x;
    if (n >= NN) return;
    float dn = den2[n];
    float v[7], m = -3.402823466e38f;
    #pragma unroll
    for (int c = 0; c < 7; ++c) {
        v[c] = acc2[(size_t)n * 8 + c] / dn + b2[c];
        m = fmaxf(m, v[c]);
    }
    float s = 0.f;
    #pragma unroll
    for (int c = 0; c < 7; ++c) s += __expf(v[c] - m);
    float ls = logf(s);
    #pragma unroll
    for (int c = 0; c < 7; ++c) out[(size_t)n * 7 + c] = v[c] - m - ls;
}

// ---------------------------------------------------------------------------
extern "C" void kernel_launch(void* const* d_in, const int* in_sizes, int n_in,
                              void* d_out, int out_size, void* d_ws, size_t ws_size,
                              hipStream_t stream) {
    const float*     x    = (const float*)d_in[0];
    const long long* ei   = (const long long*)d_in[1];
    const float*     W1   = (const float*)d_in[2];
    const float*     at1s = (const float*)d_in[3];
    const float*     at1d = (const float*)d_in[4];
    const float*     b1   = (const float*)d_in[5];
    const float*     W2   = (const float*)d_in[6];
    const float*     at2s = (const float*)d_in[7];
    const float*     at2d = (const float*)d_in[8];
    const float*     b2   = (const float*)d_in[9];
    float* out = (float*)d_out;
    char*  ws  = (char*)d_ws;

    unsigned short* w1bf = (unsigned short*)(ws + OFF_W1BF);
    unsigned short* w2bf = (unsigned short*)(ws + OFF_W2BF);
    float*    h1   = (float*)(ws + OFF_H1);
    float*    aS1  = (float*)(ws + OFF_AS1);
    float*    aD1  = (float*)(ws + OFF_AD1);
    unsigned* max1 = (unsigned*)(ws + OFF_MAX1);
    float*    den1 = (float*)(ws + OFF_DEN1);
    float*    acc1 = (float*)(ws + OFF_ACC1);
    float*    helu = (float*)(ws + OFF_HE);
    float*    h2p  = (float*)(ws + OFF_H2);
    float*    aS2  = (float*)(ws + OFF_AS2);
    float*    aD2  = (float*)(ws + OFF_AD2);
    unsigned* max2 = (unsigned*)(ws + OFF_MAX2);
    float*    den2 = (float*)(ws + OFF_DEN2);
    float*    acc2 = (float*)(ws + OFF_ACC2);

    const unsigned KEY_NEG_INF = 0x007FFFFFu;   // fkey(-inf)
    const int B = 256;
    auto grid = [](int n) { return (n + 255) / 256; };

    // weights -> bf16 (once per call; deterministic)
    k_conv_w1<<<grid(64 * KPAD), B, 0, stream>>>(W1, w1bf);
    k_conv_w2<<<grid(16 * 64),   B, 0, stream>>>(W2, w2bf);

    // init accumulators
    k_fill<<<grid(NN * 8),  B, 0, stream>>>(max1, KEY_NEG_INF, NN * 8);
    k_fill<<<grid(NN * 8),  B, 0, stream>>>((unsigned*)den1, 0u, NN * 8);
    k_fill<<<grid(NN * 64), B, 0, stream>>>((unsigned*)acc1, 0u, NN * 64);
    k_fill<<<grid(NN),      B, 0, stream>>>(max2, KEY_NEG_INF, NN);
    k_fill<<<grid(NN),      B, 0, stream>>>((unsigned*)den2, 0u, NN);
    k_fill<<<grid(NN * 8),  B, 0, stream>>>((unsigned*)acc2, 0u, NN * 8);

    // ---- layer 1 ----
    k_gemm1<<<NN / 32, B, 0, stream>>>(x, w1bf, h1);
    k_attdot1<<<grid(NN * 8), B, 0, stream>>>(h1, at1s, at1d, aS1, aD1);
    k_edge_max<8><<<grid(ENTOT), B, 0, stream>>>(ei, aS1, aD1, max1);
    k_edge_acc<8, 8, 64><<<grid(ENTOT * 8), B, 0, stream>>>(ei, aS1, aD1, max1, den1, h1, acc1);
    k_final1<<<grid(NN * 64), B, 0, stream>>>(acc1, den1, b1, helu);

    // ---- layer 2 ----
    k_gemm2<<<(NN / 16 + 7) / 8, B, 0, stream>>>(helu, w2bf, h2p);
    k_attdot2<<<grid(NN), B, 0, stream>>>(h2p, at2s, at2d, aS2, aD2);
    k_edge_max<1><<<grid(ENTOT), B, 0, stream>>>(ei, aS2, aD2, max2);
    k_edge_acc<1, 7, 8><<<grid(ENTOT), B, 0, stream>>>(ei, aS2, aD2, max2, den2, h2p, acc2);
    k_final2<<<grid(NN), B, 0, stream>>>(acc2, den2, b2, out);
}
